// Grasp_62543313764361
// MI455X (gfx1250) — matrix operand order repaired and checked by reading, not compiled, not run on hardware
//
#include <hip/hip_runtime.h>
#include <hip/hip_bf16.h>

#define NNODE 384
#define NEDGE 3456
#define NLAB  32
#define F3    64
#define KD    32

typedef __attribute__((ext_vector_type(16))) _Float16 v16h;
typedef __attribute__((ext_vector_type(8)))  float    v8f;

__device__ __forceinline__ v8f zero8() {
  v8f r;
#pragma unroll
  for (int i = 0; i < 8; ++i) r[i] = 0.f;
  return r;
}

// D = A(16x32 f16) * B(32x16 f16) + C(16x16 f32)
__device__ __forceinline__ v8f wmma16(v16h a, v16h b, v8f c) {
  return __builtin_amdgcn_wmma_f32_16x16x32_f16(false, a, false, b, (short)0, c,
                                                false, false);
}

// A-operand loader from row-major [rows][ld] f16. Row = lane&15.
// lane<16 holds K in {kb..kb+15}; lane>=16 holds {kb+16..kb+31}.
__device__ __forceinline__ v16h load_A16(const _Float16* base, int ld, int kb, int lane) {
  const int hl = lane >> 4;
  const _Float16* p = base + (lane & 15) * ld + kb + (hl ? 16 : 0);
  v16h a;
#pragma unroll
  for (int e = 0; e < 8; ++e) a[e] = p[e];
#pragma unroll
  for (int e = 0; e < 8; ++e) a[8 + e] = p[8 + e];
  return a;
}

// 16 contiguous halves (two b128 loads).
__device__ __forceinline__ v16h load_contig16(const _Float16* p) {
  v16h b;
#pragma unroll
  for (int e = 0; e < 16; ++e) b[e] = p[e];
  return b;
}

// B-operand from a lane-swizzled weight buffer: buf[(nb*32 + lane)*16 + e].
__device__ __forceinline__ v16h load_Bsw(const _Float16* buf, int nb, int lane) {
  return load_contig16(buf + (nb * 32 + lane) * 16);
}

__device__ __forceinline__ float sigm(float x) { return 1.f / (1.f + expf(-x)); }

// ---------------------------------------------------------------------------
// Kernel 1: GIN conv pass for both graphs (1 block per graph, 384 threads).
// Produces hs16[g][l][node][c] as f16 (6 layers per graph).
// ---------------------------------------------------------------------------
__global__ void __launch_bounds__(384)
conv_kernel(const float* __restrict__ f1, const float* __restrict__ f2,
            const int* __restrict__ ei1, const int* __restrict__ ei2,
            const float* __restrict__ l0_w1, const float* __restrict__ l0_b1,
            const float* __restrict__ l0_w2, const float* __restrict__ l0_b2,
            const float* __restrict__ gin_eps,
            const float* __restrict__ gin_w1, const float* __restrict__ gin_b1,
            const float* __restrict__ gin_w2, const float* __restrict__ gin_b2,
            const float* __restrict__ gin_gamma, const float* __restrict__ gin_beta,
            _Float16* __restrict__ hs16) {
  __shared__ float sh_h[NNODE * F3];    // 98304 B
  __shared__ float sh_agg[NNODE * 32];  // 49152 B (also reused for adjacency bits)
  __shared__ float sh_bn[2 * F3];

  const int g = blockIdx.x;
  const float* feats = g ? f2 : f1;
  const int* ei = g ? ei2 : ei1;
  _Float16* hsg = hs16 + (size_t)g * 6 * NNODE * F3;
  const int tid = threadIdx.x;  // node id

  // ---- degree = row-popcount of distinct-edge adjacency (self-loops excluded)
  unsigned* adj = (unsigned*)sh_agg;  // NNODE*12 words = 18432 B
  for (int idx = tid; idx < NNODE * 12; idx += NNODE) adj[idx] = 0u;
  __syncthreads();
  for (int e = tid; e < NEDGE - NNODE; e += NNODE) {
    int s = ei[e], d = ei[NEDGE + e];
    atomicOr(&adj[s * 12 + (d >> 5)], 1u << (d & 31));
  }
  __syncthreads();
  float deg = 0.f;
  for (int w2 = 0; w2 < 12; ++w2) deg += (float)__popc(adj[tid * 12 + w2]);
  __syncthreads();  // adj region freed for agg use

  // ---- layer0: lin(relu(lin([feats, deg])))
  float z1[F3], z[F3];
  for (int c = 0; c < F3; ++c) {
    float acc = l0_b1[c];
    for (int d = 0; d < NLAB; ++d) acc += feats[tid * NLAB + d] * l0_w1[d * F3 + c];
    acc += deg * l0_w1[NLAB * F3 + c];
    z1[c] = fmaxf(acc, 0.f);
  }
  for (int c = 0; c < F3; ++c) {
    float acc = l0_b2[c];
    for (int d = 0; d < F3; ++d) acc += z1[d] * l0_w2[d * F3 + c];
    z[c] = acc;
  }
  for (int c = 0; c < F3; ++c) {
    sh_h[tid * F3 + c] = z[c];           // h = h0 (no relu)
    hsg[tid * F3 + c] = (_Float16)z[c];  // hs[0] = h0
  }
  __syncthreads();

  // ---- 5 GIN layers
  for (int li = 0; li < 5; ++li) {
    float zin[F3];
    const float epsv = 1.f + gin_eps[li];
    for (int half = 0; half < 2; ++half) {  // channel halves to bound LDS
      const int cb = half * 32;
      for (int idx = tid; idx < NNODE * 32; idx += NNODE) sh_agg[idx] = 0.f;
      __syncthreads();
      for (int e = tid; e < NEDGE; e += NNODE) {  // all edges incl. dups+loops
        int s = ei[e], d = ei[NEDGE + e];
        for (int c = 0; c < 32; ++c)
          atomicAdd(&sh_agg[d * 32 + c], sh_h[s * F3 + cb + c]);
      }
      __syncthreads();
      for (int c = 0; c < 32; ++c)
        zin[cb + c] = epsv * sh_h[tid * F3 + cb + c] + sh_agg[tid * 32 + c];
      __syncthreads();
    }
    const float* w1 = gin_w1 + li * F3 * F3;
    const float* b1 = gin_b1 + li * F3;
    const float* w2p = gin_w2 + li * F3 * F3;
    const float* b2 = gin_b2 + li * F3;
    for (int c = 0; c < F3; ++c) {
      float acc = b1[c];
      for (int d = 0; d < F3; ++d) acc += zin[d] * w1[d * F3 + c];
      z1[c] = fmaxf(acc, 0.f);
    }
    for (int c = 0; c < F3; ++c) {
      float acc = b2[c];
      for (int d = 0; d < F3; ++d) acc += z1[d] * w2p[d * F3 + c];
      z[c] = acc;
    }
    // BatchNorm (batch stats)
    for (int idx = tid; idx < 2 * F3; idx += NNODE) sh_bn[idx] = 0.f;
    __syncthreads();
    for (int c = 0; c < F3; ++c) {
      atomicAdd(&sh_bn[c], z[c]);
      atomicAdd(&sh_bn[F3 + c], z[c] * z[c]);
    }
    __syncthreads();
    for (int c = 0; c < F3; ++c) {
      const float mu = sh_bn[c] * (1.f / NNODE);
      const float var = sh_bn[F3 + c] * (1.f / NNODE) - mu * mu;
      z[c] = (z[c] - mu) * rsqrtf(var + 1e-5f) * gin_gamma[li * F3 + c] +
             gin_beta[li * F3 + c];
    }
    for (int c = 0; c < F3; ++c)
      hsg[(size_t)(li + 1) * NNODE * F3 + tid * F3 + c] = (_Float16)z[c];
    __syncthreads();
    for (int c = 0; c < F3; ++c) sh_h[tid * F3 + c] = fmaxf(z[c], 0.f);
    __syncthreads();
  }
}

// ---------------------------------------------------------------------------
// Kernel 2: t[l,k] = h1[l] (384x64) @ ged[l,k] (64x64), f16 out, via WMMA.
// ged staged to LDS pre-swizzled into B-operand lane order.
// ---------------------------------------------------------------------------
__global__ void __launch_bounds__(256)
tmat_kernel(const _Float16* __restrict__ hs16, const float* __restrict__ ged,
            _Float16* __restrict__ t16) {
  __shared__ __align__(32) _Float16 s_gm[4 * 32 * 2 * 16];  // [ct][lane][chunk][e]
  const int lk = blockIdx.x;  // 0..191
  const int l = lk >> 5;
  const float* gm = ged + (size_t)lk * 64 * 64;
  for (int idx = threadIdx.x; idx < 4096; idx += 256) {
    const int e = idx & 15, ch = (idx >> 4) & 1, ls = (idx >> 5) & 31, ct = idx >> 10;
    const int kidx = ch * 32 + e + 16 * (ls >> 4);
    const int col = ct * 16 + (ls & 15);
    s_gm[idx] = (_Float16)gm[kidx * 64 + col];
  }
  __syncthreads();

  const _Float16* h1 = hs16 + (size_t)l * NNODE * F3;  // graph 0
  const int w = threadIdx.x >> 5, lane = threadIdx.x & 31;
  const int hl = lane >> 4, nn = lane & 15;
  for (int job = w; job < 96; job += 8) {
    const int it = job >> 2, ct = job & 3;  // 24 row-tiles x 4 col-chunks
    v8f acc = zero8();
#pragma unroll
    for (int ch = 0; ch < 2; ++ch) {
      v16h a = load_A16(h1 + (size_t)(it * 16) * F3, F3, ch * 32, lane);
      v16h b = load_contig16(s_gm + ((ct * 32 + lane) * 2 + ch) * 16);
      acc = wmma16(a, b, acc);
    }
    _Float16* dst = t16 + ((size_t)lk * NNODE + it * 16) * F3 + ct * 16 + nn;
#pragma unroll
    for (int r = 0; r < 8; ++r) dst[(size_t)(r + 8 * hl) * F3] = (_Float16)acc[r];
  }
}

// ---------------------------------------------------------------------------
// Kernel 3: fused per-(16x16) tile: stack tiles -> GRU -> attention -> q ->
// cost matrix softmax -> context. 256 threads (8 waves), all GEMMs on WMMA.
// Pair index p = i_local*16 + j_local; wave w owns pairs [32w, 32w+32).
// ---------------------------------------------------------------------------
__global__ void __launch_bounds__(256)
fused_kernel(const _Float16* __restrict__ t16, const _Float16* __restrict__ hs16,
             const float* __restrict__ wih, const float* __restrict__ whh,
             const float* __restrict__ bih, const float* __restrict__ bhh,
             const float* __restrict__ wattn_w, const float* __restrict__ wattn_b,
             const float* __restrict__ v_attn,
             const float* __restrict__ wq1_w, const float* __restrict__ wq1_b,
             const float* __restrict__ wq2_w, const float* __restrict__ wq2_b,
             const float* __restrict__ wk_w, const float* __restrict__ wk_b,
             const float* __restrict__ wv_w, const float* __restrict__ wv_b,
             float* __restrict__ out) {
  __shared__ __align__(32) _Float16 s_stack[6 * 256 * KD];  // 98304 B stack[l][p][k]
  __shared__ __align__(32) _Float16 s_out[6 * 256 * KD];    // 98304 B GRU outputs
  __shared__ __align__(32) _Float16 s_h[256 * KD];          // 16384 B GRU hidden
  __shared__ __align__(32) _Float16 s_q[256 * KD];          // 16384 B staging/pattern/q
  __shared__ __align__(32) _Float16 s_wih[6 * 32 * 16];     // 6144 B swizzled wih^T
  __shared__ __align__(32) _Float16 s_whh[6 * 32 * 16];     // 6144 B swizzled whh^T
  __shared__ __align__(32) _Float16 s_wsm[5][2 * 32 * 16];  // 10240 B swizzled 32x32 w
  __shared__ float s_sc[6 * 256];                           // 6144 B

  const int tid = threadIdx.x;
  const int w = tid >> 5, lane = tid & 31;
  const int hl = lane >> 4, nn = lane & 15;
  const int i0 = blockIdx.y * 16, j0 = blockIdx.x * 16;

  // ---- stage weights into B-operand lane-swizzled order + zero hidden
  // s[(nb*32+ls)*16+e] = W_T[k = e + 16*(ls>>4)][col = nb*16 + (ls&15)]
  for (int idx = tid; idx < 6 * 512; idx += 256) {
    const int e = idx & 15, ls = (idx >> 4) & 31, nb = idx >> 9;
    const int k = e + 16 * (ls >> 4);
    const int c = nb * 16 + (ls & 15);
    s_wih[idx] = (_Float16)wih[c * KD + k];  // wih (96,32): wih^T[k][c]
    s_whh[idx] = (_Float16)whh[c * KD + k];
  }
  for (int idx = tid; idx < 1024; idx += 256) {
    const int e = idx & 15, ls = (idx >> 4) & 31, nb = idx >> 9;
    const int k = e + 16 * (ls >> 4);
    const int c = nb * 16 + (ls & 15);
    const int src = k * KD + c;  // row-major (K_in, K_out)
    s_wsm[0][idx] = (_Float16)wattn_w[src];
    s_wsm[1][idx] = (_Float16)wq1_w[src];
    s_wsm[2][idx] = (_Float16)wq2_w[src];
    s_wsm[3][idx] = (_Float16)wk_w[src];
    s_wsm[4][idx] = (_Float16)wv_w[src];
  }
  for (int idx = tid; idx < 256 * KD; idx += 256) s_h[idx] = (_Float16)0.f;

  // ---- Stage A: stack[l][p][k] = (t[l,k] @ h2[l]^T) tile, K=64 -> 2 WMMA
  const _Float16* h2 = hs16 + (size_t)6 * NNODE * F3;  // graph 1
  for (int job = w; job < 192; job += 8) {             // job = l*32 + k
    const int l = job >> 5, k = job & 31;
    v8f acc = zero8();
    const _Float16* abase = t16 + ((size_t)job * NNODE + i0) * F3;
    const _Float16* brow = h2 + (size_t)l * NNODE * F3 + (size_t)(j0 + nn) * F3;
#pragma unroll
    for (int ch = 0; ch < 2; ++ch) {
      v16h a = load_A16(abase, F3, ch * 32, lane);
      v16h b = load_contig16(brow + ch * 32 + 16 * hl);
      acc = wmma16(a, b, acc);
    }
#pragma unroll
    for (int r = 0; r < 8; ++r) {
      const int p = (r + 8 * hl) * 16 + nn;  // i_local*16 + j_local
      s_stack[(l * 256 + p) * KD + k] = (_Float16)acc[r];
    }
  }
  __syncthreads();  // below is wave-local (wave owns pairs 32w..32w+31)

  // ---- Stage B: GRU; hoist all 12 B operands into registers once
  {
    v16h Bih[6], Bhh[6];
#pragma unroll
    for (int q = 0; q < 6; ++q) {
      Bih[q] = load_Bsw(s_wih, q, lane);
      Bhh[q] = load_Bsw(s_whh, q, lane);
    }
    for (int l = 0; l < 6; ++l) {
      for (int mi = 0; mi < 2; ++mi) {
        const int mc = 2 * w + mi;  // 16-pair chunk
        v16h ax = load_A16(s_stack + (l * 256 + mc * 16) * KD, KD, 0, lane);
        v16h ah = load_A16(s_h + mc * 16 * KD, KD, 0, lane);
#pragma unroll
        for (int nb2 = 0; nb2 < 2; ++nb2) {
          v8f gr = wmma16(ax, Bih[nb2], zero8());
          v8f gz = wmma16(ax, Bih[nb2 + 2], zero8());
          v8f gn = wmma16(ax, Bih[nb2 + 4], zero8());
          v8f hr = wmma16(ah, Bhh[nb2], zero8());
          v8f hz = wmma16(ah, Bhh[nb2 + 2], zero8());
          v8f hn = wmma16(ah, Bhh[nb2 + 4], zero8());
          const int c = nb2 * 16 + nn;
          const float biR = bih[c], bhR = bhh[c];
          const float biZ = bih[32 + c], bhZ = bhh[32 + c];
          const float biN = bih[64 + c], bhN = bhh[64 + c];
#pragma unroll
          for (int r = 0; r < 8; ++r) {
            const int p = mc * 16 + r + 8 * hl;
            const float rr = sigm((gr[r] + biR) + (hr[r] + bhR));
            const float zz = sigm((gz[r] + biZ) + (hz[r] + bhZ));
            const float nv = tanhf((gn[r] + biN) + rr * (hn[r] + bhN));
            const float hold = (float)s_h[p * KD + c];
            const float hnew = (1.f - zz) * nv + zz * hold;
            s_h[p * KD + c] = (_Float16)hnew;
            s_out[(l * 256 + p) * KD + c] = (_Float16)hnew;
          }
        }
      }
    }
  }

  // ---- Stage C: attention over layers (per pair), pattern
  const int myp = w * 32 + lane;  // this lane's own pair
  float myscore[6];
  for (int l = 0; l < 6; ++l) {
    for (int mi = 0; mi < 2; ++mi) {
      const int mc = 2 * w + mi;
      v16h ao = load_A16(s_out + (l * 256 + mc * 16) * KD, KD, 0, lane);
#pragma unroll
      for (int nb = 0; nb < 2; ++nb) {
        v8f u = wmma16(ao, load_Bsw(s_wsm[0], nb, lane), zero8());
        const float bb = wattn_b[nb * 16 + nn];
#pragma unroll
        for (int r = 0; r < 8; ++r) {
          const int p = mc * 16 + r + 8 * hl;
          s_q[p * KD + nb * 16 + nn] = (_Float16)tanhf(u[r] + bb);
        }
      }
    }
    float sc = 0.f;
    for (int c = 0; c < KD; ++c) sc += (float)s_q[myp * KD + c] * v_attn[c];
    myscore[l] = sc;
  }
  {  // softmax over layers + pattern -> s_q
    float mx = myscore[0];
    for (int l = 1; l < 6; ++l) mx = fmaxf(mx, myscore[l]);
    float ssum = 0.f;
    for (int l = 0; l < 6; ++l) { myscore[l] = expf(myscore[l] - mx); ssum += myscore[l]; }
    const float inv = 1.f / ssum;
    for (int c = 0; c < KD; ++c) {
      float acc = 0.f;
      for (int l = 0; l < 6; ++l)
        acc += myscore[l] * inv * (float)s_out[(l * 256 + myp) * KD + c];
      s_q[myp * KD + c] = (_Float16)acc;  // pattern
    }
  }

  // ---- Stage D: q = relu(pattern@wq1+b)@wq2+b
  for (int mi = 0; mi < 2; ++mi) {
    const int mc = 2 * w + mi;
    v16h ap = load_A16(s_q + mc * 16 * KD, KD, 0, lane);
#pragma unroll
    for (int nb = 0; nb < 2; ++nb) {
      v8f u = wmma16(ap, load_Bsw(s_wsm[1], nb, lane), zero8());
      const float bb = wq1_b[nb * 16 + nn];
#pragma unroll
      for (int r = 0; r < 8; ++r)
        s_h[(mc * 16 + r + 8 * hl) * KD + nb * 16 + nn] =
            (_Float16)fmaxf(u[r] + bb, 0.f);
    }
  }
  for (int mi = 0; mi < 2; ++mi) {
    const int mc = 2 * w + mi;
    v16h ap = load_A16(s_h + mc * 16 * KD, KD, 0, lane);
#pragma unroll
    for (int nb = 0; nb < 2; ++nb) {
      v8f u = wmma16(ap, load_Bsw(s_wsm[2], nb, lane), zero8());
      const float bb = wq2_b[nb * 16 + nn];
#pragma unroll
      for (int r = 0; r < 8; ++r)
        s_q[(mc * 16 + r + 8 * hl) * KD + nb * 16 + nn] = (_Float16)(u[r] + bb);  // q
    }
  }

  // ---- Stage E: kk per layer -> cost s -> softmax over l -> context
  for (int l = 0; l < 6; ++l) {
    for (int mi = 0; mi < 2; ++mi) {
      const int mc = 2 * w + mi;
      v16h as = load_A16(s_stack + (l * 256 + mc * 16) * KD, KD, 0, lane);
      v8f k0 = wmma16(as, load_Bsw(s_wsm[3], 0, lane), zero8());
      v8f k1 = wmma16(as, load_Bsw(s_wsm[3], 1, lane), zero8());
      const float b0 = wk_b[nn], b1v = wk_b[16 + nn];
#pragma unroll
      for (int r = 0; r < 8; ++r) {
        const int p = mc * 16 + r + 8 * hl;
        float part = (k0[r] + b0) * (float)s_q[p * KD + nn] +
                     (k1[r] + b1v) * (float)s_q[p * KD + 16 + nn];
        part += __shfl_xor(part, 1, 16);
        part += __shfl_xor(part, 2, 16);
        part += __shfl_xor(part, 4, 16);
        part += __shfl_xor(part, 8, 16);
        if (nn == 0) s_sc[l * 256 + p] = part;
      }
    }
  }
  {  // per-pair softmax over layers (scale 1/sqrt(32))
    float sv[6];
    float mx = -1e30f;
    for (int l = 0; l < 6; ++l) {
      sv[l] = s_sc[l * 256 + myp] * 0.17677669529663687f;
      mx = fmaxf(mx, sv[l]);
    }
    float s2 = 0.f;
    for (int l = 0; l < 6; ++l) { sv[l] = expf(sv[l] - mx); s2 += sv[l]; }
    const float in2 = 1.f / s2;
    for (int l = 0; l < 6; ++l) s_sc[l * 256 + myp] = sv[l] * in2;
  }
  for (int mi = 0; mi < 2; ++mi) {
    const int mc = 2 * w + mi;
    v8f c0 = zero8(), c1 = zero8();
    const float b0 = wv_b[nn], b1v = wv_b[16 + nn];
    for (int l = 0; l < 6; ++l) {
      v16h as = load_A16(s_stack + (l * 256 + mc * 16) * KD, KD, 0, lane);
      v8f u0 = wmma16(as, load_Bsw(s_wsm[4], 0, lane), zero8());
      v8f u1 = wmma16(as, load_Bsw(s_wsm[4], 1, lane), zero8());
#pragma unroll
      for (int r = 0; r < 8; ++r) {
        const float al = s_sc[l * 256 + mc * 16 + r + 8 * hl];
        c0[r] += al * (u0[r] + b0);
        c1[r] += al * (u1[r] + b1v);
      }
    }
#pragma unroll
    for (int r = 0; r < 8; ++r) {
      const size_t i = (size_t)(i0 + mc);
      const size_t j = (size_t)(j0 + r + 8 * hl);
      out[(i * NNODE + j) * KD + nn] = c0[r];
      out[(i * NNODE + j) * KD + 16 + nn] = c1[r];
    }
  }
}

// ---------------------------------------------------------------------------
extern "C" void kernel_launch(void* const* d_in, const int* in_sizes, int n_in,
                              void* d_out, int out_size, void* d_ws, size_t ws_size,
                              hipStream_t stream) {
  const float* f1 = (const float*)d_in[0];
  const float* f2 = (const float*)d_in[1];
  const float* l0_w1 = (const float*)d_in[2];
  const float* l0_b1 = (const float*)d_in[3];
  const float* l0_w2 = (const float*)d_in[4];
  const float* l0_b2 = (const float*)d_in[5];
  const float* gin_eps = (const float*)d_in[6];
  const float* gin_w1 = (const float*)d_in[7];
  const float* gin_b1 = (const float*)d_in[8];
  const float* gin_w2 = (const float*)d_in[9];
  const float* gin_b2 = (const float*)d_in[10];
  const float* gin_gamma = (const float*)d_in[11];
  const float* gin_beta = (const float*)d_in[12];
  const float* ged_mats = (const float*)d_in[13];
  const float* gru_wih = (const float*)d_in[14];
  const float* gru_whh = (const float*)d_in[15];
  const float* gru_bih = (const float*)d_in[16];
  const float* gru_bhh = (const float*)d_in[17];
  const float* wattn_w = (const float*)d_in[18];
  const float* wattn_b = (const float*)d_in[19];
  const float* v_attn = (const float*)d_in[20];
  const float* wq1_w = (const float*)d_in[21];
  const float* wq1_b = (const float*)d_in[22];
  const float* wq2_w = (const float*)d_in[23];
  const float* wq2_b = (const float*)d_in[24];
  const float* wk_w = (const float*)d_in[25];
  const float* wk_b = (const float*)d_in[26];
  const float* wv_w = (const float*)d_in[27];
  const float* wv_b = (const float*)d_in[28];
  const int* ei1 = (const int*)d_in[29];
  const int* ei2 = (const int*)d_in[30];
  float* out = (float*)d_out;

  // workspace layout (f16): hs16 [2][6][384][64], then t16 [192][384][64]
  _Float16* hs16 = (_Float16*)d_ws;
  _Float16* t16 = (_Float16*)((char*)d_ws + (size_t)2 * 6 * NNODE * F3 * 2);

  conv_kernel<<<2, 384, 0, stream>>>(f1, f2, ei1, ei2, l0_w1, l0_b1, l0_w2, l0_b2,
                                     gin_eps, gin_w1, gin_b1, gin_w2, gin_b2,
                                     gin_gamma, gin_beta, hs16);
  tmat_kernel<<<192, 256, 0, stream>>>(hs16, ged_mats, t16);
  fused_kernel<<<dim3(24, 24), 256, 0, stream>>>(
      t16, hs16, gru_wih, gru_whh, gru_bih, gru_bhh, wattn_w, wattn_b, v_attn,
      wq1_w, wq1_b, wq2_w, wq2_b, wk_w, wk_b, wv_w, wv_b, out);
}
